// BiLSTM_9852654977128
// MI455X (gfx1250) — matrix operand order repaired and checked by reading, not compiled, not run on hardware
//
#include <hip/hip_runtime.h>
#include <stdint.h>

#define B_   256
#define T_   512
#define DIN_ 40
#define H_   128
#define G_   512   // 4*H
#define NC_  7

typedef __attribute__((ext_vector_type(16))) __bf16 v16bf_t;
typedef __attribute__((ext_vector_type(8)))  float  v8f_t;

union Frag16 { v16bf_t v; uint32_t u[8]; };

static __device__ __forceinline__ uint32_t pack_bf16(float a, float b) {
  uint32_t ua = __float_as_uint(a);
  uint32_t ub = __float_as_uint(b);
  ua += 0x7fffu + ((ua >> 16) & 1u);   // round-to-nearest-even
  ub += 0x7fffu + ((ub >> 16) & 1u);
  return (ua >> 16) | (ub & 0xffff0000u);
}
static __device__ __forceinline__ uint16_t cvt_bf16(float a) {
  uint32_t ua = __float_as_uint(a);
  ua += 0x7fffu + ((ua >> 16) & 1u);
  return (uint16_t)(ua >> 16);
}
static __device__ __forceinline__ float sigf(float x) {
  float cx = fminf(fmaxf(x, -30.f), 30.f);
  return 1.f / (1.f + __expf(-cx));
}
static __device__ __forceinline__ float tanh_f(float x) {
  float cx = fminf(fmaxf(x, -15.f), 15.f);
  float e = __expf(2.f * cx);
  return (e - 1.f) / (e + 1.f);
}

// ---------------------------------------------------------------------------
// fp32 (M x Ks) row-major -> bf16 (M x Kd) row-major, zero-padded, dword packed
// ---------------------------------------------------------------------------
__global__ void cvt_pad_bf16(const float* __restrict__ src, uint32_t* __restrict__ dst,
                             int M, int Ks, int Kd) {
  int idx = blockIdx.x * blockDim.x + threadIdx.x;       // dword index
  int kdw = Kd >> 1;
  if (idx >= M * kdw) return;
  int row = idx / kdw;
  int k0  = (idx - row * kdw) * 2;
  float a = (k0     < Ks) ? src[(size_t)row * Ks + k0    ] : 0.f;
  float b = (k0 + 1 < Ks) ? src[(size_t)row * Ks + k0 + 1] : 0.f;
  dst[idx] = pack_bf16(a, b);
}

// ---------------------------------------------------------------------------
// pre = A(bf16, (B*T) x K) * W(bf16, 512 x K)^T + bias, written directly in
// WMMA C-fragment layout:  frag[(bt*T + t)*32 + ntile][lane][8 floats]
// grid = (T, 8 n-groups, 4 bt-groups), 128 threads (4 waves).
// Wave = batch-tile of 16 rows b = bt*16+rloc at fixed t  (row index b*T+t),
// so each fragment matches exactly what lstm_scan consumes.
// ---------------------------------------------------------------------------
__global__ __launch_bounds__(128)
void gemm_pre(const uint32_t* __restrict__ A, const uint32_t* __restrict__ W,
              const float* __restrict__ bias, float* __restrict__ Cfrag, int K) {
  const int lane = threadIdx.x & 31;
  const int wave = threadIdx.x >> 5;
  const int t    = blockIdx.x;
  const int n0   = blockIdx.y * 64;
  const int bt   = blockIdx.z * 4 + wave;     // batch tile 0..15
  const int rloc = lane & 15;
  const int hi   = lane >> 4;
  const int kdw  = K >> 1;

  const uint32_t* arow = A + ((size_t)(bt * 16 + rloc) * T_ + t) * kdw;

  v8f_t acc[4];
  #pragma unroll
  for (int tl = 0; tl < 4; ++tl)
    #pragma unroll
    for (int r = 0; r < 8; ++r) acc[tl][r] = 0.f;

  for (int kb = 0; kb < K; kb += 32) {
    Frag16 af;
    const int kdwb = kb >> 1;
    #pragma unroll
    for (int v = 0; v < 4; ++v) af.u[v]     = arow[kdwb + hi * 8 + v];
    #pragma unroll
    for (int v = 0; v < 4; ++v) af.u[4 + v] = arow[kdwb + hi * 8 + 4 + v];
    #pragma unroll
    for (int tl = 0; tl < 4; ++tl) {
      Frag16 bf;
      const uint32_t* wrow = W + (size_t)(n0 + tl * 16 + rloc) * kdw;
      #pragma unroll
      for (int v = 0; v < 8; ++v) bf.u[v] = wrow[kdwb + hi * 8 + v];
      acc[tl] = __builtin_amdgcn_wmma_f32_16x16x32_bf16(
          false, af.v, false, bf.v, (short)0, acc[tl], false, false);
    }
  }
  #pragma unroll
  for (int tl = 0; tl < 4; ++tl) {
    float bv = bias[n0 + tl * 16 + rloc];     // C-fragment lane owns column n
    float4* dst = (float4*)(Cfrag +
        ((((size_t)bt * T_ + t) * 32 + (blockIdx.y * 4 + tl)) * 32 + lane) * 8);
    float4 lo = { acc[tl][0] + bv, acc[tl][1] + bv, acc[tl][2] + bv, acc[tl][3] + bv };
    float4 hh = { acc[tl][4] + bv, acc[tl][5] + bv, acc[tl][6] + bv, acc[tl][7] + bv };
    dst[0] = lo;
    dst[1] = hh;
  }
}

// ---------------------------------------------------------------------------
// Bidirectional LSTM scan.  grid = (16 batch-tiles, 2 dirs), 256 threads
// (8 waves).  Wave w owns j-columns [16w,16w+16) of all 4 gates (n-tiles
// w, w+8, w+16, w+24).  h in LDS (bf16), c in registers, W_hh fragments in
// registers.  pre fragments are contiguous per lane (2 x b128 per gate),
// double-buffered in registers one step ahead + global_prefetch two ahead.
// ---------------------------------------------------------------------------
__global__ __launch_bounds__(256)
void lstm_scan(const float* __restrict__ preF, const float* __restrict__ preR,
               const float* __restrict__ whhF, const float* __restrict__ whhR,
               uint32_t* __restrict__ outdw /* (B,T,256) bf16 = 128 dwords */) {
  const int dir   = blockIdx.y;
  const float* pre = dir ? preR : preF;
  const float* whh = dir ? whhR : whhF;
  const int bbase = blockIdx.x * 16;
  const int lane  = threadIdx.x & 31;
  const int wave  = threadIdx.x >> 5;    // j-tile 0..7
  const int nloc  = lane & 15;
  const int hi    = lane >> 4;

  __shared__ uint32_t hlds[16 * 64];     // 16 x 128 bf16

  // W_hh fragments [gate][kstep]; B layout: K = kb + hi*16 + 2v (+1), N = nloc
  Frag16 wf[4][4];
  #pragma unroll
  for (int g = 0; g < 4; ++g) {
    const float* wr = whh + (size_t)((wave + 8 * g) * 16 + nloc) * H_;
    #pragma unroll
    for (int ks = 0; ks < 4; ++ks)
      #pragma unroll
      for (int v = 0; v < 8; ++v) {
        int k0 = ks * 32 + hi * 16 + 2 * v;
        wf[g][ks].u[v] = pack_bf16(wr[k0], wr[k0 + 1]);
      }
  }
  for (int i = threadIdx.x; i < 16 * 64; i += 256) hlds[i] = 0;

  v8f_t cst;
  #pragma unroll
  for (int r = 0; r < 8; ++r) cst[r] = 0.f;

  // fragment base for this (batch-tile); tile index = (bt*T + t)*32 + ntile
  const float* fragbase = pre + (size_t)blockIdx.x * T_ * 32 * 256;

  auto load_frag = [&](int t, v8f_t (&dstv)[4]) {
    #pragma unroll
    for (int g = 0; g < 4; ++g) {
      const float4* p = (const float4*)(fragbase +
          (((size_t)t * 32) + (wave + 8 * g)) * 256 + lane * 8);
      float4 lo = p[0], hh = p[1];
      dstv[g][0] = lo.x; dstv[g][1] = lo.y; dstv[g][2] = lo.z; dstv[g][3] = lo.w;
      dstv[g][4] = hh.x; dstv[g][5] = hh.y; dstv[g][6] = hh.z; dstv[g][7] = hh.w;
    }
  };

  v8f_t accA[4], accB[4];
  load_frag(dir ? (T_ - 1) : 0, accA);
  __syncthreads();

  auto step = [&](int st, v8f_t (&acc)[4], v8f_t (&nxt)[4]) {
    const int t = dir ? (T_ - 1 - st) : st;

    // gates += h @ W_hh^T   (K = 128 in 4 steps of 32)
    #pragma unroll
    for (int ks = 0; ks < 4; ++ks) {
      Frag16 af;                         // A: M = nloc, K = kb + hi*16 + s (contiguous, matches B)
      const uint32_t* hrow = hlds + nloc * 64;
      #pragma unroll
      for (int v = 0; v < 4; ++v) af.u[v]     = hrow[ks * 16 + hi * 8 + v];
      #pragma unroll
      for (int v = 0; v < 4; ++v) af.u[4 + v] = hrow[ks * 16 + hi * 8 + 4 + v];
      #pragma unroll
      for (int g = 0; g < 4; ++g)
        acc[g] = __builtin_amdgcn_wmma_f32_16x16x32_bf16(
            false, af.v, false, wf[g][ks].v, (short)0, acc[g], false, false);
    }

    // prefetch next step (registers) + step after next (L2)
    if (st + 1 < T_) load_frag(dir ? (T_ - 2 - st) : (st + 1), nxt);
    if (st + 2 < T_) {
      int t2 = dir ? (T_ - 3 - st) : (st + 2);
      __builtin_prefetch(fragbase + (((size_t)t2 * 32) + wave) * 256 + lane * 8, 0, 0);
    }
    __syncthreads();   // all A-frag reads of hlds complete

    // activations + state update (lane-local)
    const int j = wave * 16 + nloc;
    #pragma unroll
    for (int r = 0; r < 8; ++r) {
      float iv = sigf(acc[0][r]);
      float fv = sigf(acc[1][r]);
      float gv = tanh_f(acc[2][r]);
      float ov = sigf(acc[3][r]);
      float cn = fv * cst[r] + iv * gv;
      cst[r] = cn;
      float hv = ov * tanh_f(cn);
      ((uint16_t*)hlds)[(r + hi * 8) * 128 + j] = cvt_bf16(hv);
    }
    __syncthreads();   // new h visible

    // coalesced bf16 h store: one ds_load_b128 + global_store_b128 per thread
    {
      const int m  = threadIdx.x >> 4;        // row 0..15
      const int qd = (threadIdx.x & 15) * 4;  // dword offset 0..60
      uint4 d = *(const uint4*)&hlds[m * 64 + qd];
      *(uint4*)&outdw[((size_t)(bbase + m) * T_ + t) * 128 + dir * 64 + qd] = d;
    }
  };

  for (int st = 0; st < T_; st += 2) {
    step(st,     accA, accB);
    step(st + 1, accB, accA);
  }
}

// ---------------------------------------------------------------------------
// out[b][c] = h2[b, T-1, :] . fc_w[c] + fc_b[c]
// ---------------------------------------------------------------------------
__global__ void fc_kernel(const uint16_t* __restrict__ h2,
                          const float* __restrict__ fcw,
                          const float* __restrict__ fcb,
                          float* __restrict__ out) {
  int b = threadIdx.x;
  if (b >= B_) return;
  const uint16_t* hrow = h2 + ((size_t)b * T_ + (T_ - 1)) * 256;
  for (int c = 0; c < NC_; ++c) {
    float s = fcb[c];
    for (int jj = 0; jj < 256; ++jj) {
      float hv = __uint_as_float(((uint32_t)hrow[jj]) << 16);
      s += hv * fcw[c * 256 + jj];
    }
    out[b * NC_ + c] = s;
  }
}

extern "C" void kernel_launch(void* const* d_in, const int* in_sizes, int n_in,
                              void* d_out, int out_size, void* d_ws, size_t ws_size,
                              hipStream_t stream) {
  (void)in_sizes; (void)n_in; (void)out_size; (void)ws_size;
  const float* x     = (const float*)d_in[0];
  const float* wih0f = (const float*)d_in[1];
  const float* whh0f = (const float*)d_in[2];
  const float* b0f   = (const float*)d_in[3];
  const float* wih0r = (const float*)d_in[4];
  const float* whh0r = (const float*)d_in[5];
  const float* b0r   = (const float*)d_in[6];
  const float* wih1f = (const float*)d_in[7];
  const float* whh1f = (const float*)d_in[8];
  const float* b1f   = (const float*)d_in[9];
  const float* wih1r = (const float*)d_in[10];
  const float* whh1r = (const float*)d_in[11];
  const float* b1r   = (const float*)d_in[12];
  const float* fcw   = (const float*)d_in[13];
  const float* fcb   = (const float*)d_in[14];

  uint8_t* ws = (uint8_t*)d_ws;
  uint32_t* xpad  = (uint32_t*)(ws + 0);            // 131072 x 64 bf16  (16.78 MB)
  uint32_t* wl0f  = (uint32_t*)(ws + 16777216);     // 512 x 64 bf16
  uint32_t* wl0r  = (uint32_t*)(ws + 16842752);
  uint32_t* wl1f  = (uint32_t*)(ws + 16908288);     // 512 x 256 bf16
  uint32_t* wl1r  = (uint32_t*)(ws + 17170432);
  uint32_t* h1seq = (uint32_t*)(ws + 17432576);     // (B,T,256) bf16  (67.1 MB)
  uint32_t* h2seq = (uint32_t*)(ws + 84541440);     // (B,T,256) bf16
  float*    pref  = (float*)   (ws + 151650304);    // frag (B*T,512) f32 (268 MB, reused)
  float*    prer  = (float*)   (ws + 420085760);    // frag (B*T,512) f32 (268 MB, reused)

  const int BT = B_ * T_;

  // bf16 conversion / K-padding
  int dw = BT * 32;
  cvt_pad_bf16<<<(dw + 255) / 256, 256, 0, stream>>>(x, xpad, BT, DIN_, 64);
  dw = 512 * 32;
  cvt_pad_bf16<<<(dw + 255) / 256, 256, 0, stream>>>(wih0f, wl0f, 512, DIN_, 64);
  cvt_pad_bf16<<<(dw + 255) / 256, 256, 0, stream>>>(wih0r, wl0r, 512, DIN_, 64);
  dw = 512 * 128;
  cvt_pad_bf16<<<(dw + 255) / 256, 256, 0, stream>>>(wih1f, wl1f, 512, 256, 256);
  cvt_pad_bf16<<<(dw + 255) / 256, 256, 0, stream>>>(wih1r, wl1r, 512, 256, 256);

  dim3 gg(T_, G_ / 64, B_ / 64);

  // layer 0
  gemm_pre<<<gg, 128, 0, stream>>>(xpad, wl0f, b0f, pref, 64);
  gemm_pre<<<gg, 128, 0, stream>>>(xpad, wl0r, b0r, prer, 64);
  lstm_scan<<<dim3(B_ / 16, 2), 256, 0, stream>>>(pref, prer, whh0f, whh0r, h1seq);
  // layer 1 (pre buffers reused)
  gemm_pre<<<gg, 128, 0, stream>>>(h1seq, wl1f, b1f, pref, 256);
  gemm_pre<<<gg, 128, 0, stream>>>(h1seq, wl1r, b1r, prer, 256);
  lstm_scan<<<dim3(B_ / 16, 2), 256, 0, stream>>>(pref, prer, whh1f, whh1r, h2seq);
  // classifier
  fc_kernel<<<1, 256, 0, stream>>>((const uint16_t*)h2seq, fcw, fcb, (float*)d_out);
}
